// MyDNNDecoder_45423574123080
// MI455X (gfx1250) — matrix operand order repaired and checked by reading, not compiled, not run on hardware
//
#include <hip/hip_runtime.h>
#include <hip/hip_bf16.h>
#include <math.h>
#include <stdint.h>

// ---------------------------------------------------------------------------
// MI455X (gfx1250) fused implementation.
//  Kernel 1: one workgroup (256 thr = 8 wave32) per batch sample.
//    TDM tensor_load_to_lds of enc -> attention (WMMA f16) ->
//    conv1/bn/relu (WMMA implicit GEMM) -> maxpool -> 16 resblocks
//    (WMMA, fused BN+residual+ReLU) -> avgpool -> feat[b][512] in workspace.
//    All intermediates live in a 136KB LDS arena (2 WGs/WGP).
//  Kernel 2: FC GEMM 1024x512x180 (padded 192) with WMMA + sigmoid epilogue.
// ---------------------------------------------------------------------------

typedef _Float16 h16;
typedef __attribute__((ext_vector_type(16))) _Float16 v16h;
typedef __attribute__((ext_vector_type(8)))  float    v8f;
typedef __attribute__((ext_vector_type(4)))  unsigned int u32x4;
typedef __attribute__((ext_vector_type(8)))  int      i32x8;
typedef __attribute__((ext_vector_type(4)))  int      i32x4;

#define BN_EPS 1e-5f
#define TLEN 144
#define DMODEL 64
#define NTHR 256
#define NWAVE 8

// Tensor Data Mover availability (per-pass: device pass sees the builtin)
#ifdef __has_builtin
#if __has_builtin(__builtin_amdgcn_tensor_load_to_lds)
#define HAVE_TDM 1
#endif
#endif
#ifndef HAVE_TDM
#define HAVE_TDM 0
#endif

// LDS arena layout (bytes)
#define OFF_ENCH    0        // [160][64] f16 (20480 B); later cbuf [64][144] f16
#define OFF_SCORES  20480    // TDM staging [144][64] f32 (36864 B) during stage 1;
                             // then [144][160] f32 scores; whh f16 in place (pitch 320 h)
                             // later: bufs0/1/2 (3 x 32768 B) at 20480/53248/86016
#define OFF_PROJH   112640   // [144][64] f16 (18432 B); later x0buf [64][160] f16
#define OFF_WH      131072   // [64][64] f16 (8192 B)
#define ARENA_BYTES 139264

struct KParams {
  const float* enc;
  const float* attn_W;
  const float* conv1_w; const float* conv1_b;
  const float* bn0g; const float* bn0b; const float* bn0rm; const float* bn0rv;
  const float* c1w[16];
  const float* bn1g[16]; const float* bn1b[16]; const float* bn1rm[16]; const float* bn1rv[16];
  const float* c2w[16];
  const float* bn2g[16]; const float* bn2b[16]; const float* bn2rm[16]; const float* bn2rv[16];
  const float* scw[16];
  const float* scg[16]; const float* scb[16]; const float* scrm[16]; const float* scrv[16];
  const float* fcw; const float* fcb;
  float* feat;
};

// Block specs (ResNet-ish): 16 blocks
__device__ const int g_INCH[16] = {64,64,64, 64,128,128,128, 128,256,256,256,256,256, 256,128,128};
__device__ const int g_OUTCH[16]= {64,64,64, 128,128,128,128, 256,256,256,256,256,256, 128,128,128};
__device__ const int g_STR[16]  = {1,1,1, 2,1,1,1, 2,1,1,1,1,1, 2,1,1};
__device__ const int g_LIN[16]  = {46,46,46, 46,23,23,23, 23,12,12,12,12,12, 12,6,6};

__device__ __forceinline__ v8f wmma16(v16h a, v16h b, v8f c) {
  return __builtin_amdgcn_wmma_f32_16x16x32_f16(false, a, false, b, (short)0, c, false, false);
}
__device__ __forceinline__ v8f zero8() { v8f z = {0.f,0.f,0.f,0.f,0.f,0.f,0.f,0.f}; return z; }

// A fragment from row-major f16 in LDS: A(m,k) = base[(row0+m)*pitch + k0+k]
__device__ __forceinline__ v16h load_a_lds(const h16* base, int row0, int k0, int pitch) {
  int lane = threadIdx.x & 31; int m = lane & 15; int hi = lane >> 4;
  const h16* r = base + (row0 + m) * pitch + k0 + hi * 16;
  v16h a;
#pragma unroll
  for (int i = 0; i < 8; ++i) a[i] = r[i];
#pragma unroll
  for (int i = 0; i < 8; ++i) a[8 + i] = r[8 + i];
  return a;
}

// B fragment from row-major f16 in LDS: B(k,n) = base[(k0+k)*pitch + col0+n]
__device__ __forceinline__ v16h load_b_lds(const h16* base, int k0, int col0, int pitch) {
  int lane = threadIdx.x & 31; int n = lane & 15; int hi = lane >> 4;
  const h16* r = base + (k0 + hi * 16) * pitch + col0 + n;
  v16h b;
#pragma unroll
  for (int i = 0; i < 16; ++i) b[i] = r[i * pitch];
  return b;
}

// Transposed B: B(k,n) = base[(col0+n)*pitch + k0+k]   (scores: enc^T)
__device__ __forceinline__ v16h load_bt_lds(const h16* base, int k0, int col0, int pitch) {
  int lane = threadIdx.x & 31; int n = lane & 15; int hi = lane >> 4;
  const h16* r = base + (col0 + n) * pitch + k0 + hi * 16;
  v16h b;
#pragma unroll
  for (int i = 0; i < 16; ++i) b[i] = r[i];
  return b;
}

// A fragment from f32 conv weights in global (L2-resident):
//   A(m,k) = w[((row0+m)*kTot + k0+k)*ksz + tap]   (f32 -> f16 on the fly)
__device__ __forceinline__ v16h load_a_w(const float* w, int row0, int kTot,
                                         int k0, int ksz, int tap) {
  int lane = threadIdx.x & 31; int m = lane & 15; int hi = lane >> 4;
  const float* r = w + ((size_t)(row0 + m) * kTot + k0 + hi * 16) * ksz + tap;
  v16h a;
#pragma unroll
  for (int i = 0; i < 8; ++i) a[i] = (h16)r[(size_t)i * ksz];
#pragma unroll
  for (int i = 0; i < 8; ++i) a[8 + i] = (h16)r[(size_t)(8 + i) * ksz];
  return a;
}

// B fragment for conv from LDS activations: B(k,n) = x[(k0+k)*pitch + (t0+n)*stride + tap]
__device__ __forceinline__ v16h load_b_conv(const h16* x, int k0, int t0,
                                            int stride, int tap, int pitch) {
  int lane = threadIdx.x & 31; int n = lane & 15; int hi = lane >> 4;
  const h16* r = x + (k0 + hi * 16) * pitch + (t0 + n) * stride + tap;
  v16h b;
#pragma unroll
  for (int i = 0; i < 16; ++i) b[i] = r[i * pitch];
  return b;
}

// Implicit-GEMM conv + folded BN (+optional residual, +optional ReLU).
// x points so that x[ci*pitchX + t*stride + tap] is input position t*stride+tap-pad.
__device__ void conv_bn(const h16* x, int ciTot, int pitchX, int stride, int ksz,
                        const float* w, const float* scale, const float* shift,
                        h16* y, int pitchY, int haloY, int Lout, int co,
                        const h16* res, int pitchR, int haloR, bool relu) {
  const int lane = threadIdx.x & 31;
  const int wave = threadIdx.x >> 5;
  const int hi = lane >> 4;
  const int nl = lane & 15;
  const int coT = co >> 4;
  const int tT = (Lout + 15) >> 4;
  for (int tile = wave; tile < coT * tT; tile += NWAVE) {
    int ct = tile / tT, tt = tile % tT;
    int co0 = ct * 16, t0 = tt * 16;
    v8f acc = zero8();
    for (int tap = 0; tap < ksz; ++tap)
      for (int k0 = 0; k0 < ciTot; k0 += 32) {
        v16h a = load_a_w(w, co0, ciTot, k0, ksz, tap);
        v16h b = load_b_conv(x, k0, t0, stride, tap, pitchX);
        acc = wmma16(a, b, acc);
      }
    int t = t0 + nl;
    if (t < Lout) {
#pragma unroll
      for (int r = 0; r < 8; ++r) {
        int ch = co0 + hi * 8 + r;
        float v = acc[r] * scale[ch] + shift[ch];
        if (res) v += (float)res[ch * pitchR + haloR + t];
        if (relu) v = fmaxf(v, 0.f);
        y[ch * pitchY + haloY + t] = (h16)v;
      }
    }
  }
}

__global__ __launch_bounds__(NTHR)
void fused_attn_cnn(KParams P) {
  extern __shared__ char arena[];
  __shared__ float sc_s[256];
  __shared__ float sh_s[256];

  const int tid = threadIdx.x;
  const int lane = tid & 31;
  const int wave = tid >> 5;
  const int b = blockIdx.x;

  h16*   ench    = (h16*)(arena + OFF_ENCH);     // [160][64], rows 144..159 = 0
  float* scoresf = (float*)(arena + OFF_SCORES); // [144][160] f32
  h16*   whh     = (h16*)(arena + OFF_SCORES);   // softmax weights, pitch 320 h
  h16*   projh   = (h16*)(arena + OFF_PROJH);    // [144][64]
  h16*   Wh      = (h16*)(arena + OFF_WH);       // [64][64]
  h16*   x0buf   = (h16*)(arena + OFF_PROJH);    // [64][160], halo 2 (reuse)
  h16*   cbuf    = (h16*)(arena + OFF_ENCH);     // [64][144]  (reuse)
  h16*   bufs0   = (h16*)(arena + 20480);        // [256][64]
  h16*   bufs1   = (h16*)(arena + 53248);
  h16*   bufs2   = (h16*)(arena + 86016);

  // ---- stage 1: enc -> LDS (TDM DMA) -> f16; W -> f16 ----------------------
  const float* enc = P.enc + (size_t)b * (TLEN * DMODEL);
#if HAVE_TDM
  {
    float* stage = (float*)(arena + OFF_SCORES);   // f32 staging (region free now)
    if (wave == 0) {
      // Tensor DMA descriptor: 1-D copy of 9216 f32 elements, global -> LDS.
      unsigned lds_addr = (unsigned)(uintptr_t)stage;          // flat low 32 = LDS addr
      unsigned long long ga = (unsigned long long)(uintptr_t)enc;
      u32x4 g0;
      g0[0] = 1u;                                              // count=1, load, no gather
      g0[1] = lds_addr;                                        // lds_addr
      g0[2] = (unsigned)(ga & 0xFFFFFFFFu);                    // global_addr[31:0]
      g0[3] = (unsigned)((ga >> 32) & 0x1FFFFFFu) | (2u << 30);// addr[56:32] | type=2
      i32x8 g1;
      g1[0] = (int)(2u << 16);                       // data_size = 4B; no multicast
      g1[1] = (int)((9216u & 0xFFFFu) << 16);        // tensor_dim0[15:0]
      g1[2] = (int)((9216u >> 16) | (1u << 16));     // tensor_dim0[31:16] | tensor_dim1 lo
      g1[3] = (int)(9216u << 16);                    // tensor_dim1 hi | tile_dim0=9216
      g1[4] = 1;                                     // tile_dim1=1, tile_dim2=0
      g1[5] = 9216;                                  // tensor_dim0_stride lo32
      g1[6] = 0;
      g1[7] = 0;
      i32x4 z4 = {0, 0, 0, 0};
#if defined(__clang_major__) && (__clang_major__ >= 23)
      i32x8 z8 = {0, 0, 0, 0, 0, 0, 0, 0};
      __builtin_amdgcn_tensor_load_to_lds(g0, g1, z4, z4, z8, 0);
#else
      __builtin_amdgcn_tensor_load_to_lds(g0, g1, z4, z4, 0);
#endif
      __builtin_amdgcn_s_wait_tensorcnt(0);
    }
    __syncthreads();
    for (int i = tid; i < 160 * 64; i += NTHR)
      ench[i] = (i < TLEN * DMODEL) ? (h16)stage[i] : (h16)0.f;
  }
#else
  for (int i = tid; i < 160 * 64; i += NTHR) {
    int r = i >> 6;
    ench[i] = (r < TLEN) ? (h16)enc[r * DMODEL + (i & 63)] : (h16)0.f;
  }
#endif
  for (int i = tid; i < 64 * 64; i += NTHR) Wh[i] = (h16)P.attn_W[i];
  __builtin_prefetch(P.conv1_w, 0, 1);   // global_prefetch_b8: warm conv1 weights in L2
  __syncthreads();

  // ---- stage 2: proj = enc @ W  (9x4 tiles, K=64) --------------------------
  for (int tile = wave; tile < 36; tile += NWAVE) {
    int mi = tile >> 2, ni = tile & 3;
    v8f acc = zero8();
    for (int k0 = 0; k0 < 64; k0 += 32) {
      v16h a = load_a_lds(ench, mi * 16, k0, 64);
      v16h bb = load_b_lds(Wh, k0, ni * 16, 64);
      acc = wmma16(a, bb, acc);
    }
    int m0 = (lane >> 4) * 8, n = lane & 15;
#pragma unroll
    for (int r = 0; r < 8; ++r)
      projh[(mi * 16 + m0 + r) * 64 + ni * 16 + n] = (h16)acc[r];
  }
  __syncthreads();

  // ---- stage 3: scores = proj @ enc^T  (9x9 tiles, K=64) -------------------
  for (int tile = wave; tile < 81; tile += NWAVE) {
    int mi = tile / 9, si = tile % 9;
    v8f acc = zero8();
    for (int k0 = 0; k0 < 64; k0 += 32) {
      v16h a = load_a_lds(projh, mi * 16, k0, 64);
      v16h bb = load_bt_lds(ench, k0, si * 16, 64);
      acc = wmma16(a, bb, acc);
    }
    int m0 = (lane >> 4) * 8, n = lane & 15;
#pragma unroll
    for (int r = 0; r < 8; ++r)
      scoresf[(mi * 16 + m0 + r) * 160 + si * 16 + n] = acc[r];
  }
  __syncthreads();

  // ---- stage 4: row softmax, f16 in place (pitch 320 h); zero x0buf --------
  for (int t = tid; t < TLEN; t += NTHR) {
    float* row = scoresf + t * 160;
    float mx = row[0];
    for (int s = 1; s < TLEN; ++s) mx = fmaxf(mx, row[s]);
    float sum = 0.f;
    for (int s = 0; s < TLEN; ++s) { float e = __expf(row[s] - mx); row[s] = e; sum += e; }
    float inv = 1.f / sum;
    h16* hr = (h16*)row;                   // f16 row fits under f32 row
    for (int s = 0; s < TLEN; ++s) hr[s] = (h16)(row[s] * inv);
    for (int s = TLEN; s < 160; ++s) hr[s] = (h16)0.f;  // pad K to 160
  }
  for (int i = tid; i < 64 * 160; i += NTHR) x0buf[i] = (h16)0.f;
  __syncthreads();

  // ---- stage 5: context = w @ enc (9x4 tiles, K=160) -> x0buf transposed ---
  for (int tile = wave; tile < 36; tile += NWAVE) {
    int mi = tile >> 2, ni = tile & 3;
    v8f acc = zero8();
    for (int k0 = 0; k0 < 160; k0 += 32) {
      v16h a = load_a_lds(whh, mi * 16, k0, 320);
      v16h bb = load_b_lds(ench, k0, ni * 16, 64);
      acc = wmma16(a, bb, acc);
    }
    int m0 = (lane >> 4) * 8, n = lane & 15;
#pragma unroll
    for (int r = 0; r < 8; ++r) {
      int t = mi * 16 + m0 + r;
      int d = ni * 16 + n;
      x0buf[d * 160 + 2 + t] = (h16)acc[r];   // [channel][time], halo 2
    }
  }
  __syncthreads();

  // ---- stage 6: conv1(k9,p2) + bias + bn0 + relu -> cbuf[64][144] ----------
  for (int i = tid; i < 64; i += NTHR) {
    float s = P.bn0g[i] * rsqrtf(P.bn0rv[i] + BN_EPS);
    sc_s[i] = s;
    sh_s[i] = (P.conv1_b[i] - P.bn0rm[i]) * s + P.bn0b[i];
  }
  __syncthreads();
  conv_bn(x0buf, 64, 160, 1, 9, P.conv1_w, sc_s, sh_s,
          cbuf, 144, 0, 140, 64, nullptr, 0, 0, true);
  __syncthreads();

  // ---- maxpool(3,3) -> bufs0 [64][64] halo 1 -------------------------------
  for (int i = tid; i < 64 * 64; i += NTHR) bufs0[i] = (h16)0.f;
  __syncthreads();
  for (int i = tid; i < 64 * 46; i += NTHR) {
    int c = i / 46, t = i % 46;
    const h16* cr = cbuf + c * 144 + t * 3;
    float m = fmaxf(fmaxf((float)cr[0], (float)cr[1]), (float)cr[2]);
    bufs0[c * 64 + 1 + t] = (h16)m;
  }
  __syncthreads();

  // ---- 16 resblocks --------------------------------------------------------
  h16* bufarr[3] = {bufs0, bufs1, bufs2};
  int cur = 0, tmp = 1, spare = 2;
  for (int blk = 0; blk < 16; ++blk) {
    int inch = g_INCH[blk], ch = g_OUTCH[blk], st = g_STR[blk], Lin = g_LIN[blk];
    int Lout = (st == 2) ? ((Lin - 1) / 2 + 1) : Lin;
    bool hassc = (P.scw[blk] != nullptr);
    h16* X = bufarr[cur];
    h16* T = bufarr[tmp];

    if (blk < 15) __builtin_prefetch(P.c1w[blk + 1], 0, 1);  // stay ahead in L2

    // conv1: X -> T  (relu(bn1(.)))
    for (int i = tid; i < ch; i += NTHR) {
      float s = P.bn1g[blk][i] * rsqrtf(P.bn1rv[blk][i] + BN_EPS);
      sc_s[i] = s; sh_s[i] = P.bn1b[blk][i] - P.bn1rm[blk][i] * s;
    }
    for (int i = tid; i < ch * 64; i += NTHR) T[i] = (h16)0.f;
    __syncthreads();
    conv_bn(X, inch, 64, st, 3, P.c1w[blk], sc_s, sh_s,
            T, 64, 1, Lout, ch, nullptr, 0, 0, true);
    __syncthreads();

    h16* R;
    if (hassc) {                 // projection shortcut: X -> spare
      h16* S = bufarr[spare];
      for (int i = tid; i < ch; i += NTHR) {
        float s = P.scg[blk][i] * rsqrtf(P.scrv[blk][i] + BN_EPS);
        sc_s[i] = s; sh_s[i] = P.scb[blk][i] - P.scrm[blk][i] * s;
      }
      for (int i = tid; i < ch * 64; i += NTHR) S[i] = (h16)0.f;
      __syncthreads();
      conv_bn(X + 1 /*halo1, pad0*/, inch, 64, st, 1, P.scw[blk], sc_s, sh_s,
              S, 64, 1, Lout, ch, nullptr, 0, 0, false);
      __syncthreads();
      R = S;
    } else {
      R = X;                     // identity shortcut
    }

    // conv2: T -> R  (relu(bn2(.) + R)), in-place residual (lane-private RMW)
    for (int i = tid; i < ch; i += NTHR) {
      float s = P.bn2g[blk][i] * rsqrtf(P.bn2rv[blk][i] + BN_EPS);
      sc_s[i] = s; sh_s[i] = P.bn2b[blk][i] - P.bn2rm[blk][i] * s;
    }
    __syncthreads();
    conv_bn(T, ch, 64, 1, 3, P.c2w[blk], sc_s, sh_s,
            R, 64, 1, Lout, ch, R, 64, 1, true);
    __syncthreads();

    if (hassc) { int t2 = cur; cur = spare; spare = t2; }
  }

  // ---- avgpool(3, s=1): [128][6] -> [128][4] -> feat[b][512] ---------------
  {
    h16* X = bufarr[cur];
    float* feat = P.feat + (size_t)b * 512;
    for (int i = tid; i < 512; i += NTHR) {
      int c = i >> 2, j = i & 3;
      const h16* r = X + c * 64 + 1 + j;
      feat[i] = ((float)r[0] + (float)r[1] + (float)r[2]) * (1.f / 3.f);
    }
  }
}

// ---- FC: out = sigmoid(feat[1024x512] @ fc_w[512x180] + fc_b) --------------
__global__ __launch_bounds__(NTHR)
void fc_sigmoid(KParams P, float* out) {
  extern __shared__ char smem[];
  h16* bh = (h16*)smem;                 // [32][192] f16 chunk of fc_w
  const int tid = threadIdx.x, lane = tid & 31, wave = tid >> 5;
  const int row0 = (blockIdx.x * NWAVE + wave) * 16;

  v8f acc[12];
#pragma unroll
  for (int i = 0; i < 12; ++i) acc[i] = zero8();

  __builtin_prefetch(P.fcw, 0, 1);

  for (int kc = 0; kc < 16; ++kc) {     // K = 512 in chunks of 32
    for (int i = tid; i < 32 * 192; i += NTHR) {
      int k = i / 192, n = i % 192;
      bh[i] = (n < 180) ? (h16)P.fcw[(size_t)(kc * 32 + k) * 180 + n] : (h16)0.f;
    }
    __syncthreads();
    v16h a = load_a_w(P.feat, row0, 512, kc * 32, 1, 0);   // f32 feat -> f16 frag
#pragma unroll
    for (int nt = 0; nt < 12; ++nt) {
      v16h bb = load_b_lds(bh, 0, nt * 16, 192);
      acc[nt] = wmma16(a, bb, acc[nt]);
    }
    __syncthreads();
  }

  int hi = lane >> 4, nl = lane & 15;
  for (int nt = 0; nt < 12; ++nt) {
    int n = nt * 16 + nl;
    if (n >= 180) continue;
    float bias = P.fcb[n];
#pragma unroll
    for (int r = 0; r < 8; ++r) {
      int row = row0 + hi * 8 + r;
      float v = acc[nt][r] + bias;
      out[(size_t)row * 180 + n] = 1.f / (1.f + __expf(-v));
    }
  }
}

extern "C" void kernel_launch(void* const* d_in, const int* in_sizes, int n_in,
                              void* d_out, int out_size, void* d_ws, size_t ws_size,
                              hipStream_t stream) {
  (void)n_in; (void)out_size; (void)ws_size;
  KParams P;
#define FPTR(i) ((const float*)d_in[(i)])
  P.enc = FPTR(0);
  // d_in[1] = edges_batch (unused by forward)
  int idx = 2;
  P.attn_W  = FPTR(idx++); P.conv1_w = FPTR(idx++); P.conv1_b = FPTR(idx++);
  P.bn0g = FPTR(idx++); P.bn0b = FPTR(idx++); P.bn0rm = FPTR(idx++); P.bn0rv = FPTR(idx++);
  for (int blk = 0; blk < 16; ++blk) {
    P.c1w[blk]  = FPTR(idx++);
    P.bn1g[blk] = FPTR(idx++); P.bn1b[blk] = FPTR(idx++);
    P.bn1rm[blk]= FPTR(idx++); P.bn1rv[blk]= FPTR(idx++);
    P.c2w[blk]  = FPTR(idx++);
    P.bn2g[blk] = FPTR(idx++); P.bn2b[blk] = FPTR(idx++);
    P.bn2rm[blk]= FPTR(idx++); P.bn2rv[blk]= FPTR(idx++);
    bool hassc = (blk == 3 || blk == 7 || blk == 13);
    if (hassc) {
      P.scw[blk] = FPTR(idx++);
      P.scg[blk] = FPTR(idx++); P.scb[blk] = FPTR(idx++);
      P.scrm[blk]= FPTR(idx++); P.scrv[blk]= FPTR(idx++);
    } else {
      P.scw[blk] = nullptr;
      P.scg[blk] = P.scb[blk] = P.scrm[blk] = P.scrv[blk] = nullptr;
    }
  }
  P.fcw = FPTR(idx++); P.fcb = FPTR(idx++);
#undef FPTR
  P.feat = (float*)d_ws;                       // [B][512] f32 scratch (2 MB @ B=1024)

  const int B = in_sizes[0] / (TLEN * DMODEL); // 1024
  fused_attn_cnn<<<B, NTHR, ARENA_BYTES, stream>>>(P);
  fc_sigmoid<<<B / (16 * NWAVE), NTHR, 32 * 192 * 2, stream>>>(P, (float*)d_out);
}
